// LatentGRUDecoderS_40690520162955
// MI455X (gfx1250) — matrix operand order repaired and checked by reading, not compiled, not run on hardware
//
#include <hip/hip_runtime.h>

#define B_   256
#define T_   512
#define H_   512
#define G3H_ 1536

typedef _Float16 v16h __attribute__((ext_vector_type(16)));
typedef _Float16 v8h  __attribute__((ext_vector_type(8)));
typedef float    v8f  __attribute__((ext_vector_type(8)));

__device__ __forceinline__ v16h cat8(v8h lo, v8h hi) {
  return __builtin_shufflevector(lo, hi, 0, 1, 2, 3, 4, 5, 6, 7,
                                         8, 9, 10, 11, 12, 13, 14, 15);
}

// Convert f32 weight matrix W[rows x K] (row-major) into f16 fragment-major
// layout for v_wmma_f32_16x16x32_f16 B-operands:
//   dst[((nt*Ksteps + ks)*32 + lane)*16 + j] = W[nt*16 + lane%16][ks*32 + (lane/16)*8 + (j%8) + (j/8)*16]
__global__ void wswz_kernel(const float* __restrict__ W, _Float16* __restrict__ dst,
                            int rows, int K) {
  int idx = blockIdx.x * blockDim.x + threadIdx.x;
  int total = rows * K;
  if (idx >= total) return;
  int j      = idx & 15;
  int lane   = (idx >> 4) & 31;
  int blk    = idx >> 9;
  int ksteps = K >> 5;
  int nt = blk / ksteps;
  int ks = blk - nt * ksteps;
  int n  = nt * 16 + (lane & 15);
  int kk = ks * 32 + ((lane >> 4) << 3) + (j & 7) + ((j >> 3) << 4);
  dst[idx] = (_Float16)W[n * K + kk];
}

// One persistent workgroup per 16-row batch tile; full T loop fused.
// Per step (3 barriers): [gate update h_t] | [merged GEMM: gates_{t+1} (GRU) +
// hid_t (MLP), shared A-fragments, + x_{t+1} prefetch] | [W2 dot + band scatter].
__launch_bounds__(512, 1)
__global__ void gru_fused_kernel(
    const float* __restrict__ dtime, const unsigned char* __restrict__ s_oh,
    const unsigned char* __restrict__ valid, const float* __restrict__ encz,
    const float* __restrict__ W_ih, const float* __restrict__ b_ih,
    const float* __restrict__ b_hh, const float* __restrict__ b1,
    const float* __restrict__ W2, const float* __restrict__ b2,
    const _Float16* __restrict__ Whh_swz, const _Float16* __restrict__ W1_swz,
    float* __restrict__ out) {
  extern __shared__ char smem[];
  float*    sh_gates = (float*)(smem);               // [16][1536] f32 (96 KB)
  float*    sh_hid   = (float*)(smem + 98304);       // [16][512]  f32 (32 KB)
  float*    sh_h     = (float*)(smem + 131072);      // [16][512]  f32 (32 KB)
  _Float16* sh_hbf   = (_Float16*)(smem + 163840);   // [16][512]  f16 (16 KB) -- WMMA A
  float*    sh_w2    = (float*)(smem + 180224);      // [512]
  float4*   sh_x     = (float4*)(smem + 182272);     // [2][16] {dt,s0,s1,valid} dbl-buffer

  const int tid  = threadIdx.x;
  const int wave = tid >> 5;
  const int lane = tid & 31;
  const int b0   = blockIdx.x * 16;

  // wave-uniform (SGPR) weight bases -> saddr-form global loads, no per-lane 64b math
  const int wuni = __builtin_amdgcn_readfirstlane(wave);
  const _Float16* __restrict__ whh_w = Whh_swz + (size_t)wuni * (6 * 16 * 512);
  const _Float16* __restrict__ w1_w  = W1_swz  + (size_t)wuni * (2 * 16 * 512);

  // ---- init: hidden state, W2 cache, zero output region ----
  for (int i = tid; i < 16 * H_; i += 512) {
    int m = i >> 9, c = i & (H_ - 1);
    float h = encz[(b0 + m) * H_ + c];
    sh_h[i]   = h;
    sh_hbf[i] = (_Float16)h;
  }
  sh_w2[tid] = W2[tid];
  for (int i = tid; i < 2 * 16 * T_; i += 512) {
    int kb = i >> 13;
    int r  = (i >> 9) & 15;
    int tc = i & (T_ - 1);
    out[kb * (B_ * T_) + (b0 + r) * T_ + tc] = 0.0f;
  }

  // t-invariant input-projection weights for this thread's hidden column
  const int col = tid;  // 0..511
  float4 wr = make_float4(W_ih[(col         ) * 3 + 0], W_ih[(col         ) * 3 + 1],
                          W_ih[(col         ) * 3 + 2], b_ih[col]);
  float4 wz = make_float4(W_ih[(col +  512) * 3 + 0], W_ih[(col +  512) * 3 + 1],
                          W_ih[(col +  512) * 3 + 2], b_ih[col + 512]);
  float4 wn = make_float4(W_ih[(col + 1024) * 3 + 0], W_ih[(col + 1024) * 3 + 1],
                          W_ih[(col + 1024) * 3 + 2], b_ih[col + 1024]);
  const float b2v = b2[0];

  const int mA   = lane & 15;          // A-fragment row (M)
  const int aoff = (lane >> 4) << 3;   // A-fragment K sub-offset: 0 or 8
  const int nl   = lane & 15;          // C/D fragment column
  const int mo   = (lane >> 4) << 3;   // C/D fragment row base: 0 or 8

  // t-invariant biases for this lane's output columns
  float bhh[6], bb1[2];
#pragma unroll
  for (int j = 0; j < 6; ++j) bhh[j] = b_hh[(wuni * 6 + j) * 16 + nl];
#pragma unroll
  for (int j = 0; j < 2; ++j) bb1[j] = b1[(wuni * 2 + j) * 16 + nl];

  // merged GEMM phase: 6 GRU tiles (gates for next step) + 2 MLP tiles (hidden
  // for current step) off one shared A-fragment per k-step; also prefetches
  // x_{t+1} into the other sh_x buffer.
  auto gemm_phase = [&](int t) {
    v8f acc[8];
#pragma unroll
    for (int j = 0; j < 8; ++j) acc[j] = (v8f)(0.0f);
#pragma unroll 4
    for (int k = 0; k < 16; ++k) {
      const _Float16* ap = sh_hbf + mA * H_ + k * 32 + aoff;
      v16h afrag = cat8(*(const v8h*)ap, *(const v8h*)(ap + 16));
#pragma unroll
      for (int j = 0; j < 6; ++j) {
        v16h bfrag = *(const v16h*)(whh_w + ((((j << 4) + k) << 9) + (lane << 4)));
        acc[j] = __builtin_amdgcn_wmma_f32_16x16x32_f16(false, afrag, false, bfrag,
                                                        (short)0, acc[j], false, false);
      }
#pragma unroll
      for (int j = 0; j < 2; ++j) {
        v16h bfrag = *(const v16h*)(w1_w + ((((j << 4) + k) << 9) + (lane << 4)));
        acc[6 + j] = __builtin_amdgcn_wmma_f32_16x16x32_f16(false, afrag, false, bfrag,
                                                            (short)0, acc[6 + j], false, false);
      }
    }
    // x prefetch for the next step (double buffer)
    if (tid < 16 && (t + 1) < T_) {
      int bi = (b0 + tid) * T_ + (t + 1);
      sh_x[((t + 1) & 1) * 16 + tid] =
          make_float4(dtime[bi], (float)s_oh[2 * bi + 0], (float)s_oh[2 * bi + 1],
                      (float)valid[bi]);
    }
    // epilogue
#pragma unroll
    for (int j = 0; j < 6; ++j) {
      int n = (wuni * 6 + j) * 16 + nl;
#pragma unroll
      for (int r = 0; r < 8; ++r)
        sh_gates[(mo + r) * G3H_ + n] = acc[j][r] + bhh[j];
    }
#pragma unroll
    for (int j = 0; j < 2; ++j) {
      int n = (wuni * 2 + j) * 16 + nl;
#pragma unroll
      for (int r = 0; r < 8; ++r)
        sh_hid[(mo + r) * H_ + n] = fmaxf(acc[6 + j][r] + bb1[j], 0.0f);
    }
  };

  __syncthreads();
  // pre-loop: gates_0 from h_init (MLP half is overwritten before use); loads x_0
  gemm_phase(-1);

  int cnt0 = 0, cnt1 = 0;  // serial_to_parallel cursors (lane0 of each wave)

  for (int t = 0; t < T_; ++t) {
    __syncthreads();  // S1: gates_t + x_t ready

    // ---- GRU gate nonlinearity + masked hidden update: h_t ----
    const float4* xb = sh_x + (t & 1) * 16;
#pragma unroll 2
    for (int m = 0; m < 16; ++m) {
      float4 xv = xb[m];
      float xr = fmaf(xv.x, wr.x, fmaf(xv.y, wr.y, fmaf(xv.z, wr.z, wr.w)));
      float xz = fmaf(xv.x, wz.x, fmaf(xv.y, wz.y, fmaf(xv.z, wz.z, wz.w)));
      float xn = fmaf(xv.x, wn.x, fmaf(xv.y, wn.y, fmaf(xv.z, wn.z, wn.w)));
      float gr = sh_gates[m * G3H_ + col];
      float gz = sh_gates[m * G3H_ + col + 512];
      float gn = sh_gates[m * G3H_ + col + 1024];
      float rg = 1.0f / (1.0f + __expf(-(xr + gr)));
      float zg = 1.0f / (1.0f + __expf(-(xz + gz)));
      float ng = tanhf(fmaf(rg, gn, xn));
      float h  = sh_h[m * H_ + col];
      float hn = fmaf(zg, h - ng, ng);          // (1-z)*n + z*h
      float ho = (xv.w > 0.5f) ? hn : h;        // masked hold
      sh_h[m * H_ + col]   = ho;
      sh_hbf[m * H_ + col] = (_Float16)ho;
    }
    __syncthreads();  // S2: h_t visible

    // ---- merged GEMM: gates_{t+1} + hid_t (+ x_{t+1} prefetch) ----
    gemm_phase(t);
    __syncthreads();  // S3: hid_t visible

    // ---- MLP layer 2 (dot with W2) + serial_to_parallel scatter; wave w owns row w ----
    float s = 0.0f;
#pragma unroll
    for (int i = 0; i < 16; ++i) {
      int hc = lane + (i << 5);
      s += sh_hid[wave * H_ + hc] * sh_w2[hc];
    }
#pragma unroll
    for (int o = 16; o > 0; o >>= 1) s += __shfl_xor(s, o, 32);
    if (lane == 0) {
      float dx = s + b2v;
      float4 xv = xb[wave];
      int b = b0 + wave;
      if (xv.y > 0.5f) out[0 * (B_ * T_) + b * T_ + cnt0++] = dx;
      if (xv.z > 0.5f) out[1 * (B_ * T_) + b * T_ + cnt1++] = dx;
    }
  }
}

extern "C" void kernel_launch(void* const* d_in, const int* in_sizes, int n_in,
                              void* d_out, int out_size, void* d_ws, size_t ws_size,
                              hipStream_t stream) {
  (void)in_sizes; (void)n_in; (void)out_size; (void)ws_size;
  const float*         dtime = (const float*)d_in[0];
  const unsigned char* s_oh  = (const unsigned char*)d_in[1];  // bool [B,T,2]
  const unsigned char* valid = (const unsigned char*)d_in[2];  // bool [B,T]
  const float*         encz  = (const float*)d_in[3];
  const float*         W_ih  = (const float*)d_in[4];
  const float*         W_hh  = (const float*)d_in[5];
  const float*         b_ih  = (const float*)d_in[6];
  const float*         b_hh  = (const float*)d_in[7];
  const float*         W1    = (const float*)d_in[8];
  const float*         b1    = (const float*)d_in[9];
  const float*         W2    = (const float*)d_in[10];
  const float*         b2    = (const float*)d_in[11];
  float*               out   = (float*)d_out;

  _Float16* Whh_swz = (_Float16*)d_ws;                         // 1536*512 f16 = 1.5 MB
  _Float16* W1_swz  = (_Float16*)((char*)d_ws + (size_t)G3H_ * H_ * sizeof(_Float16));

  // weight precondition: f32 -> f16, fragment-major swizzle
  int totWhh = G3H_ * H_;
  wswz_kernel<<<(totWhh + 511) / 512, 512, 0, stream>>>(W_hh, Whh_swz, G3H_, H_);
  int totW1 = H_ * H_;
  wswz_kernel<<<(totW1 + 511) / 512, 512, 0, stream>>>(W1, W1_swz, H_, H_);

  // 16 persistent workgroups (one per 16-row batch tile), 512 threads = 16 wave32s,
  // ~179 KB dynamic LDS (CDNA5 allows up to 320 KB/WGP)
  size_t smem = 182784;
  gru_fused_kernel<<<16, 512, smem, stream>>>(dtime, s_oh, valid, encz,
                                              W_ih, b_ih, b_hh, b1, W2, b2,
                                              Whh_swz, W1_swz, out);
}
